// GNN_VAE_7292854468773
// MI455X (gfx1250) — matrix operand order repaired and checked by reading, not compiled, not run on hardware
//
#include <hip/hip_runtime.h>
#include <hip/hip_bf16.h>

// ---- sizes (fixed by the reference) ----
#define NN 8192
#define FF 256
#define MM 128
#define EE 131072
#define NT 512            // NN/16 tiles per dim
#define NSX (NT / 4)      // 128 strips of 4 col-tiles per row
#define RPB 8             // encoder rows per block

typedef __attribute__((ext_vector_type(16))) __bf16 v16bf;
typedef __attribute__((ext_vector_type(8)))  __bf16 v8bf;
typedef __attribute__((ext_vector_type(8)))  float  v8f;

// ------------------------------------------------------------------
// 0) zero adjacency bit matrix (8 MB = 2,097,152 words)
// ------------------------------------------------------------------
__global__ void zero_adj_kernel(unsigned* __restrict__ adj, int nWords) {
    int i = blockIdx.x * blockDim.x + threadIdx.x;
    if (i < nWords) adj[i] = 0u;
}

// ------------------------------------------------------------------
// 1) encoder: 8 rows per block; W_mu/W_sig streamed once per block.
//    z stored bf16; deterministic per-block KL partial.
// ------------------------------------------------------------------
__global__ __launch_bounds__(128) void encoder_kernel(
    const float* __restrict__ x, const float* __restrict__ eps,
    const float* __restrict__ W_mu, const float* __restrict__ b_mu,
    const float* __restrict__ W_sig, const float* __restrict__ b_sig,
    __hip_bfloat16* __restrict__ z, float* __restrict__ klPart)
{
    __shared__ float xs[RPB][FF];
    __shared__ float red[128];
    const int row0 = blockIdx.x * RPB;
    const int m    = threadIdx.x;          // 0..127

    #pragma unroll
    for (int rr = 0; rr < RPB; ++rr) {
        xs[rr][m]       = x[(size_t)(row0 + rr) * FF + m];
        xs[rr][m + 128] = x[(size_t)(row0 + rr) * FF + m + 128];
    }
    __syncthreads();

    float amu[RPB] = {}, asg[RPB] = {};
    #pragma unroll 4
    for (int k = 0; k < FF; ++k) {
        const float wm = W_mu[k * MM + m];
        const float ws = W_sig[k * MM + m];
        #pragma unroll
        for (int rr = 0; rr < RPB; ++rr) {
            amu[rr] = fmaf(xs[rr][k], wm, amu[rr]);
            asg[rr] = fmaf(xs[rr][k], ws, asg[rr]);
        }
    }

    float klsum = 0.f;
    #pragma unroll
    for (int rr = 0; rr < RPB; ++rr) {
        const float mu = amu[rr] + b_mu[m];
        const float ls = asg[rr] + b_sig[m];
        const float sg = __expf(ls);                       // v_exp_f32
        const float zv = fmaf(sg, eps[(size_t)(row0 + rr) * MM + m], mu);
        z[(size_t)(row0 + rr) * MM + m] = __float2bfloat16(zv);
        klsum += 0.5f * (sg * sg + mu * mu - 1.0f) - ls;
    }

    red[m] = klsum;
    __syncthreads();
    for (int off = 64; off > 0; off >>= 1) {
        if (m < off) red[m] += red[m + off];
        __syncthreads();
    }
    if (m == 0) klPart[blockIdx.x] = red[0];
}

// ------------------------------------------------------------------
// 2) adjacency bits: atomicOr is idempotent -> exact dedup like .set(1)
// ------------------------------------------------------------------
__global__ void build_adj_kernel(const int* __restrict__ ei,
                                 unsigned* __restrict__ adj) {
    int e = blockIdx.x * blockDim.x + threadIdx.x;
    if (e >= EE) return;
    unsigned s = (unsigned)ei[e]      & (NN - 1);
    unsigned d = (unsigned)ei[EE + e] & (NN - 1);
    unsigned i1 = (s << 13) + d;
    unsigned i2 = (d << 13) + s;
    atomicOr(&adj[i1 >> 5], 1u << (i1 & 31));
    atomicOr(&adj[i2 >> 5], 1u << (i2 & 31));
}

// ------------------------------------------------------------------
// 3) one wave per 16x64 strip: A fragment reused across 4 col-tiles,
//    16x v_wmma_f32_16x16x32_bf16 per strip, fast softplus epilogue.
// ------------------------------------------------------------------
__device__ __forceinline__ float tile_epilogue(
    const v8f c, int R, int Cb, int lhalf, int l15,
    const unsigned* __restrict__ adj)
{
    float sum = 0.f;
    #pragma unroll
    for (int r = 0; r < 8; ++r) {
        const int i = R + r + (lhalf << 3);
        const int j = Cb + l15;
        if (j > i) {
            const float s = c[r];
            const unsigned idx = ((unsigned)i << 13) + (unsigned)j;
            const unsigned bit = (adj[idx >> 5] >> (idx & 31)) & 1u;
            // stable softplus with HW transcendentals: v_exp_f32 / v_log_f32
            const float sp = fmaxf(s, 0.f) + __logf(1.0f + __expf(-fabsf(s)));
            sum += sp - (bit ? s : 0.f);
        }
    }
    return sum;
}

__global__ __launch_bounds__(32) void bce_strip_kernel(
    const __bf16* __restrict__ z, const unsigned* __restrict__ adj,
    float* __restrict__ stripPart)
{
    const int bx = blockIdx.x;            // strip of 4 column tiles
    const int rt = blockIdx.y;            // row tile
    const int lane = threadIdx.x;
    const int R = rt << 4;
    const int C = bx << 6;
    if (C + 63 <= R) {                    // no j>i anywhere in strip
        if (lane == 0) stripPart[rt * NSX + bx] = 0.f;
        return;
    }

    const int lhalf = lane >> 4;
    const int l15   = lane & 15;

    const __bf16* zA  = z + (size_t)(R + l15) * MM;
    const __bf16* pB0 = z + (size_t)(C + l15) * MM;
    const __bf16* pB1 = pB0 + 16 * MM;
    const __bf16* pB2 = pB0 + 32 * MM;
    const __bf16* pB3 = pB0 + 48 * MM;

    v8f c0 = {}, c1 = {}, c2 = {}, c3 = {};
    #pragma unroll
    for (int k0 = 0; k0 < MM; k0 += 32) {
        union { v16bf v; v8bf h[2]; } a, b0, b1, b2, b3;
        // A (16x32): lane<16 -> K = k0+{0..7,16..23}; lane>=16 -> +8
        a.h[0] = *(const v8bf*)(zA + k0 + (lhalf ? 8  : 0));
        a.h[1] = *(const v8bf*)(zA + k0 + (lhalf ? 24 : 16));
        // B (32x16): lane<16 -> K = k0+{0..7,16..23}; lane>=16 -> +8
        const int d0 = k0 + (lhalf ? 8  : 0);
        const int d1 = k0 + (lhalf ? 24 : 16);
        b0.h[0] = *(const v8bf*)(pB0 + d0); b0.h[1] = *(const v8bf*)(pB0 + d1);
        b1.h[0] = *(const v8bf*)(pB1 + d0); b1.h[1] = *(const v8bf*)(pB1 + d1);
        b2.h[0] = *(const v8bf*)(pB2 + d0); b2.h[1] = *(const v8bf*)(pB2 + d1);
        b3.h[0] = *(const v8bf*)(pB3 + d0); b3.h[1] = *(const v8bf*)(pB3 + d1);
        c0 = __builtin_amdgcn_wmma_f32_16x16x32_bf16(false, a.v, false, b0.v,
                                                     (short)0, c0, false, false);
        c1 = __builtin_amdgcn_wmma_f32_16x16x32_bf16(false, a.v, false, b1.v,
                                                     (short)0, c1, false, false);
        c2 = __builtin_amdgcn_wmma_f32_16x16x32_bf16(false, a.v, false, b2.v,
                                                     (short)0, c2, false, false);
        c3 = __builtin_amdgcn_wmma_f32_16x16x32_bf16(false, a.v, false, b3.v,
                                                     (short)0, c3, false, false);
    }

    float sum = tile_epilogue(c0, R, C +  0, lhalf, l15, adj)
              + tile_epilogue(c1, R, C + 16, lhalf, l15, adj)
              + tile_epilogue(c2, R, C + 32, lhalf, l15, adj)
              + tile_epilogue(c3, R, C + 48, lhalf, l15, adj);

    #pragma unroll
    for (int off = 16; off > 0; off >>= 1)
        sum += __shfl_xor(sum, off, 32);
    if (lane == 0) stripPart[rt * NSX + bx] = sum;
}

// ------------------------------------------------------------------
// 4) final fixed-order reduction: bce + 0.001*kl
// ------------------------------------------------------------------
__global__ __launch_bounds__(256) void reduce_kernel(
    const float* __restrict__ stripPart, const float* __restrict__ klPart,
    float* __restrict__ out)
{
    __shared__ float sb[256], sk[256];
    const int t = threadIdx.x;
    float b = 0.f, k = 0.f;
    for (int i = t; i < NT * NSX; i += 256) b += stripPart[i];
    for (int i = t; i < NN / RPB; i += 256) k += klPart[i];
    sb[t] = b; sk[t] = k;
    __syncthreads();
    for (int off = 128; off > 0; off >>= 1) {
        if (t < off) { sb[t] += sb[t + off]; sk[t] += sk[t + off]; }
        __syncthreads();
    }
    if (t == 0) out[0] = sb[0] + 0.001f * sk[0];
}

// ------------------------------------------------------------------
extern "C" void kernel_launch(void* const* d_in, const int* in_sizes, int n_in,
                              void* d_out, int out_size, void* d_ws, size_t ws_size,
                              hipStream_t stream) {
    (void)in_sizes; (void)n_in; (void)out_size; (void)ws_size;
    const float* x     = (const float*)d_in[0];
    const int*   ei    = (const int*)  d_in[1];   // [2, E]
    const float* eps   = (const float*)d_in[2];
    const float* W_mu  = (const float*)d_in[3];
    const float* b_mu  = (const float*)d_in[4];
    const float* W_sig = (const float*)d_in[5];
    const float* b_sig = (const float*)d_in[6];
    float* out = (float*)d_out;

    // workspace layout
    char* ws = (char*)d_ws;
    const int ADJ_WORDS = (NN * NN) / 32;                  // 2,097,152 (8 MB)
    unsigned*        adj       = (unsigned*)ws;
    __hip_bfloat16*  z         = (__hip_bfloat16*)(ws + (size_t)ADJ_WORDS * 4); // 2 MB
    float*           stripPart = (float*)(ws + 10ull * 1024 * 1024);            // 256 KB
    float*           klPart    = stripPart + NT * NSX;                          // 4 KB

    zero_adj_kernel<<<(ADJ_WORDS + 255) / 256, 256, 0, stream>>>(adj, ADJ_WORDS);

    encoder_kernel<<<NN / RPB, 128, 0, stream>>>(x, eps, W_mu, b_mu, W_sig,
                                                 b_sig, z, klPart);

    build_adj_kernel<<<(EE + 255) / 256, 256, 0, stream>>>(ei, adj);

    bce_strip_kernel<<<dim3(NSX, NT), 32, 0, stream>>>(
        (const __bf16*)z, adj, stripPart);

    reduce_kernel<<<1, 256, 0, stream>>>(stripPart, klPart, out);
}
